// newCustomLoss_26121991095008
// MI455X (gfx1250) — matrix operand order repaired and checked by reading, not compiled, not run on hardware
//
#include <hip/hip_runtime.h>
#include <math.h>

#define NROWS 8192
#define DFEAT 512
#define EPSN  1e-12f

typedef __attribute__((ext_vector_type(8)))  _Float16 v8h;
typedef __attribute__((ext_vector_type(16))) _Float16 v16h;
typedef __attribute__((ext_vector_type(8)))  float    v8f;

// ---------------------------------------------------------------------------
// Kernel 1: L2-normalize each row of featureMatrix (f32) and store as f16.
// One block (256 threads) per row; each thread handles 2 of the 512 elements.
// ---------------------------------------------------------------------------
__global__ void __launch_bounds__(256)
normalize_rows_kernel(const float* __restrict__ feat, _Float16* __restrict__ fh) {
    const int row = blockIdx.x;
    const int t   = threadIdx.x;
    const float* r = feat + (size_t)row * DFEAT;
    const float x0 = r[t];
    const float x1 = r[t + 256];

    __shared__ float sm[256];
    sm[t] = x0 * x0 + x1 * x1;
    __syncthreads();
    for (int s = 128; s > 0; s >>= 1) {
        if (t < s) sm[t] += sm[t + s];
        __syncthreads();
    }
    const float inv = 1.0f / fmaxf(sqrtf(sm[0]), EPSN);

    _Float16* o = fh + (size_t)row * DFEAT;
    o[t]       = (_Float16)(x0 * inv);
    o[t + 256] = (_Float16)(x1 * inv);
}

// ---------------------------------------------------------------------------
// Kernel 2: fused  (f f^T) tile GEMM  +  masked MSE partial reduction.
// Grid: 64 x 64 blocks, each block = 128x128 output tile, 256 threads = 8 waves.
// Wave w owns rows [iBase + 16w, +16) x cols [jBase, +128) = 8 accumulator
// tiles of 16x16 f32 held in VGPRs (v_wmma_f32_16x16x32_f16, K-loop over 512).
// The 8192x8192 similarity matrix is never written to memory.
// ---------------------------------------------------------------------------
__global__ void __launch_bounds__(256)
cos_mse_tiles_kernel(const _Float16* __restrict__ fh,
                     const float*    __restrict__ orig,
                     float*          __restrict__ partial_sq,
                     float*          __restrict__ partial_cnt) {
    const int lane  = threadIdx.x & 31;
    const int wave  = threadIdx.x >> 5;
    const int half  = lane >> 4;   // 0: lanes 0-15, 1: lanes 16-31
    const int l16   = lane & 15;
    const int iTile = blockIdx.y * 128 + wave * 16;   // this wave's 16 rows
    const int jBase = blockIdx.x * 128;               // block's 128 cols

    v8f acc[8] = {};  // eight 16x16 f32 accumulator tiles (64 VGPRs)

    // A-matrix rows: lane = M  -> row (iTile + l16) of fh, independent of half.
    const _Float16* aRow = fh + (size_t)(iTile + l16) * DFEAT;

    for (int kk = 0; kk < DFEAT; kk += 32) {
        // A 16x32 f16, contiguous k order used by this code: VGPRs 0-7 hold K = half*16 + {0..15}
        v8h alo = *(const v8h*)(aRow + kk + half * 16);
        v8h ahi = *(const v8h*)(aRow + kk + half * 16 + 8);
        v16h a = __builtin_shufflevector(alo, ahi,
                 0, 1, 2, 3, 4, 5, 6, 7, 8, 9, 10, 11, 12, 13, 14, 15);

#pragma unroll
        for (int jj = 0; jj < 8; ++jj) {
            // B 32x16 f16, same contiguous k order as A: lane = N, VGPRs 0-7 hold K = half*16 + {0..15}.
            // B = f^T, so lane n reads row (jBase + jj*16 + n) of fh: contiguous.
            const _Float16* bRow =
                fh + (size_t)(jBase + jj * 16 + l16) * DFEAT + kk + half * 16;
            v8h blo = *(const v8h*)(bRow);
            v8h bhi = *(const v8h*)(bRow + 8);
            v16h b = __builtin_shufflevector(blo, bhi,
                     0, 1, 2, 3, 4, 5, 6, 7, 8, 9, 10, 11, 12, 13, 14, 15);

            acc[jj] = __builtin_amdgcn_wmma_f32_16x16x32_f16(
                false, a, false, b, (short)0, acc[jj], false, false);
        }
    }

    // Fused masked-MSE: C/D layout -> lane holds (M = v + 8*half, N = l16).
    float sq = 0.0f, cnt = 0.0f;
#pragma unroll
    for (int jj = 0; jj < 8; ++jj) {
#pragma unroll
        for (int v = 0; v < 8; ++v) {
            const int r = iTile + half * 8 + v;
            const int c = jBase + jj * 16 + l16;
            const float o = orig[(size_t)r * NROWS + c];
            const float d = o - acc[jj][v];
            if (o != 0.0f) { sq += d * d; cnt += 1.0f; }
        }
    }

    // Deterministic block reduction -> one partial pair per block.
    __shared__ float ssq[256];
    __shared__ float scnt[256];
    const int t = threadIdx.x;
    ssq[t] = sq; scnt[t] = cnt;
    __syncthreads();
    for (int s = 128; s > 0; s >>= 1) {
        if (t < s) { ssq[t] += ssq[t + s]; scnt[t] += scnt[t + s]; }
        __syncthreads();
    }
    if (t == 0) {
        const int bid = blockIdx.y * gridDim.x + blockIdx.x;
        partial_sq[bid]  = ssq[0];
        partial_cnt[bid] = scnt[0];
    }
}

// ---------------------------------------------------------------------------
// Kernel 3: deterministic final reduction of the 4096 block partials.
// ---------------------------------------------------------------------------
__global__ void __launch_bounds__(256)
final_reduce_kernel(const float* __restrict__ partial_sq,
                    const float* __restrict__ partial_cnt,
                    float* __restrict__ out, int nparts) {
    const int t = threadIdx.x;
    float sq = 0.0f, cnt = 0.0f;
    for (int i = t; i < nparts; i += 256) {
        sq  += partial_sq[i];
        cnt += partial_cnt[i];
    }
    __shared__ float ssq[256];
    __shared__ float scnt[256];
    ssq[t] = sq; scnt[t] = cnt;
    __syncthreads();
    for (int s = 128; s > 0; s >>= 1) {
        if (t < s) { ssq[t] += ssq[t + s]; scnt[t] += scnt[t + s]; }
        __syncthreads();
    }
    if (t == 0) out[0] = ssq[0] / scnt[0];
}

// ---------------------------------------------------------------------------
extern "C" void kernel_launch(void* const* d_in, const int* in_sizes, int n_in,
                              void* d_out, int out_size, void* d_ws, size_t ws_size,
                              hipStream_t stream) {
    // setup_inputs order: originalMatrix [8192,8192] f32, featureMatrix [8192,512] f32
    const float* orig = (const float*)d_in[0];
    const float* feat = (const float*)d_in[1];

    // Workspace layout:
    //   [0, 8 MB)            : normalized features as f16 (8192 x 512)
    //   [8 MB, +16 KB)       : 4096 per-block sum-of-squares partials
    //   [8 MB+16 KB, +16 KB) : 4096 per-block count partials
    char* ws = (char*)d_ws;
    _Float16* fh       = (_Float16*)ws;
    float* partial_sq  = (float*)(ws + (size_t)NROWS * DFEAT * sizeof(_Float16));
    float* partial_cnt = partial_sq + 4096;

    normalize_rows_kernel<<<NROWS, 256, 0, stream>>>(feat, fh);

    dim3 grid(NROWS / 128, NROWS / 128);  // 64 x 64 blocks of 128x128 tiles
    cos_mse_tiles_kernel<<<grid, 256, 0, stream>>>(fh, orig, partial_sq, partial_cnt);

    final_reduce_kernel<<<1, 256, 0, stream>>>(partial_sq, partial_cnt,
                                               (float*)d_out, 4096);
}
